// EGNN_71880572666058
// MI455X (gfx1250) — matrix operand order repaired and checked by reading, not compiled, not run on hardware
//
#include <hip/hip_runtime.h>
#include <hip/hip_bf16.h>

// ---------------- problem constants ----------------
#define BB   2
#define NN   4096
#define DD   128
#define MMO  16      // message dim
#define KKN  32      // neighbors
#define EH   514     // edge hidden = 2*(2D+1)
#define EIN  257     // edge in = 2D+1
#define EHP  544     // edge-hidden K-loop extent (multiple of 32)
#define HSTR 576     // Hi/Hj row stride (multiple of 64, pad cols zeroed)
#define NIP  160     // node_in padded width (144 -> 160)

typedef __attribute__((ext_vector_type(16))) __bf16 v16bf;
typedef __attribute__((ext_vector_type(8)))  float  v8f;

union Frag16 { unsigned int u[8]; v16bf v; };

__device__ __forceinline__ float silu_f(float x) { return x / (1.0f + __expf(-x)); }

// ============================================================================
// Kernel 0: weight prep -- transpose + pad + bf16-convert:
//   dst[n*Kp + k] = (k<K && n<N) ? src[k*ldsrc + n] : 0    (dst is B^T, padded)
// Run once per weight matrix (tiny), makes every GEMM inner loop guard-free.
// ============================================================================
__global__ void egnn_prep_bt(const float* __restrict__ src, int K, int N, int ldsrc,
                             __bf16* __restrict__ dst, int Kp, int Np)
{
    int idx = blockIdx.x * 256 + threadIdx.x;
    if (idx >= Kp * Np) return;
    int n = idx / Kp;
    int k = idx - n * Kp;
    float v = (k < K && n < N) ? src[(size_t)k * ldsrc + n] : 0.0f;
    dst[(size_t)n * Kp + k] = (__bf16)v;
}

// ============================================================================
// Kernel 1: per-node top-K (smallest masked distance) selection.
// ============================================================================
__global__ void egnn_topk(const float* __restrict__ coors,
                          const unsigned char* __restrict__ mask,
                          int* __restrict__ g_idx, float* __restrict__ g_dist)
{
    __shared__ float rank_s[NN];
    __shared__ float real_s[NN];
    __shared__ float rmin[256];
    __shared__ int   rarg[256];

    int node = blockIdx.x;                 // 0 .. B*N-1
    int b    = node >> 12;
    int i    = node & (NN - 1);
    const float* cb = coors + (size_t)b * NN * 3;
    float xi = cb[i * 3 + 0], yi = cb[i * 3 + 1], zi = cb[i * 3 + 2];
    int   mi = mask[(size_t)b * NN + i];
    int   t  = threadIdx.x;

    for (int j = t; j < NN; j += 256) {
        float dx = xi - cb[j * 3 + 0];
        float dy = yi - cb[j * 3 + 1];
        float dz = zi - cb[j * 3 + 2];
        float d  = dx * dx + dy * dy + dz * dz;
        real_s[j] = d;
        int mj = mask[(size_t)b * NN + j];
        rank_s[j] = (mi && mj) ? d : 1e5f;
    }
    __syncthreads();

    for (int kk = 0; kk < KKN; ++kk) {
        float best = 3.0e38f; int barg = NN;
        for (int j = t; j < NN; j += 256) {
            float v = rank_s[j];
            if (v < best) { best = v; barg = j; }
        }
        rmin[t] = best; rarg[t] = barg;
        __syncthreads();
        for (int s = 128; s > 0; s >>= 1) {
            if (t < s) {
                float v2 = rmin[t + s]; int a2 = rarg[t + s];
                if (v2 < rmin[t] || (v2 == rmin[t] && a2 < rarg[t])) {
                    rmin[t] = v2; rarg[t] = a2;
                }
            }
            __syncthreads();
        }
        if (t == 0) {
            int jm = rarg[0];
            g_idx[node * KKN + kk]  = jm;
            g_dist[node * KKN + kk] = real_s[jm];
            rank_s[jm] = 3.0e38f;
        }
        __syncthreads();
    }
}

// ============================================================================
// Kernel 2: bf16 WMMA GEMM with pre-transposed padded bf16 weights.
//   C = act(A@B + bias) (+ residual); pad cols [Ndim, Npad) written as 0.
// BT is B^T padded: [Npad][kp] bf16, kp == Kdim, Kdim % 32 == 0,
// Mdim % 16 == 0, grid.x = Npad/64. No LDS, no barriers, no guards in k-loop.
// ============================================================================
__global__ void egnn_gemm_bf16(const float* __restrict__ A, int lda,
                               const __bf16* __restrict__ BT, int kp,
                               float* __restrict__ C, int ldc,
                               int Mdim, int Ndim, int Kdim,
                               const float* __restrict__ bias,
                               const float* __restrict__ residual,
                               int act)
{
    int lane = threadIdx.x & 31;
    int wave = threadIdx.x >> 5;
    int row0 = blockIdx.y * 16;
    int ncol = blockIdx.x * 64 + wave * 16 + (lane & 15);
    int kau  = (lane >> 4) << 3;
    int kbu  = (lane >> 4) << 3;
    const float*  ap = A  + (size_t)(row0 + (lane & 15)) * lda;
    const __bf16* bp = BT + (size_t)ncol * kp;
    v8f acc = {};

    for (int k0 = 0; k0 < Kdim; k0 += 32) {
        float av[16];
        *(float4*)&av[0]  = *(const float4*)(ap + k0 + kau);
        *(float4*)&av[4]  = *(const float4*)(ap + k0 + kau + 4);
        *(float4*)&av[8]  = *(const float4*)(ap + k0 + kau + 16);
        *(float4*)&av[12] = *(const float4*)(ap + k0 + kau + 20);
        Frag16 fa, fb;
#pragma unroll
        for (int e = 0; e < 16; ++e) fa.v[e] = (__bf16)av[e];
        *(uint4*)&fb.u[0] = *(const uint4*)(bp + k0 + kbu);
        *(uint4*)&fb.u[4] = *(const uint4*)(bp + k0 + kbu + 16);
        acc = __builtin_amdgcn_wmma_f32_16x16x32_bf16(false, fa.v, false, fb.v,
                                                      (short)0, acc, false, false);
    }

    float bs = (bias && ncol < Ndim) ? bias[ncol] : 0.0f;
#pragma unroll
    for (int r = 0; r < 8; ++r) {
        int m = row0 + r + ((lane >> 4) << 3);
        float v = acc[r] + bs;
        if (act) v = silu_f(v);
        if (residual && ncol < Ndim) v += residual[(size_t)m * ldc + ncol];
        if (ncol >= Ndim) v = 0.0f;
        C[(size_t)m * ldc + ncol] = v;
    }
}

// ============================================================================
// Kernel 3: fused edge MLP (WMMA).
// ============================================================================
__global__ void egnn_edge(const float* __restrict__ Hi, const float* __restrict__ Hj,
                          const int* __restrict__ g_idx, const float* __restrict__ g_dist,
                          const unsigned char* __restrict__ mask,
                          const float* __restrict__ w_e1, const float* __restrict__ b_e1,
                          const float* __restrict__ w_e2, const float* __restrict__ b_e2,
                          float* __restrict__ m_out)
{
    __shared__ alignas(16) __bf16 sBt[16 * EHP];   // [out-col][k]  transposed
    __shared__ alignas(16) float  swd[EHP];
    __shared__ alignas(16) float  sb1[EHP];

    int tid = threadIdx.x;
    for (int e = tid; e < 16 * EHP; e += 256) {
        int n  = e / EHP;
        int kr = e - n * EHP;
        float v = (kr < EH) ? w_e2[kr * 16 + n] : 0.0f;
        sBt[e] = (__bf16)v;
    }
    for (int c = tid; c < EHP; c += 256) {
        swd[c] = (c < EH) ? w_e1[(size_t)(EIN - 1) * EH + c] : 0.0f;
        sb1[c] = (c < EH) ? b_e1[c] : 0.0f;
    }
    __syncthreads();

    int lane = tid & 31;
    int wave = tid >> 5;
    int tile = blockIdx.x * 8 + wave;
    int e0   = tile * 16;
    int edge = e0 + (lane & 15);
    int node = edge >> 5;
    int b    = node >> 12;
    int j    = g_idx[edge];
    float dist = g_dist[edge];
    const float4* hi4 = (const float4*)(Hi + (size_t)node * HSTR);
    const float4* hj4 = (const float4*)(Hj + ((size_t)(b * NN) + j) * HSTR);
    const float4* wd4 = (const float4*)swd;
    const float4* bb4 = (const float4*)sb1;
    int kau = (lane >> 4) << 3;
    int kbu = (lane >> 4) << 3;
    int n16 = lane & 15;
    v8f acc = {};

    for (int k0 = 0; k0 < EHP; k0 += 32) {
        int q0 = (k0 + kau) >> 2;
        float hiV[16], hjV[16], wdV[16], b1V[16];
        *(float4*)&hiV[0]  = hi4[q0];     *(float4*)&hiV[4]  = hi4[q0 + 1];
        *(float4*)&hiV[8]  = hi4[q0 + 4]; *(float4*)&hiV[12] = hi4[q0 + 5];
        *(float4*)&hjV[0]  = hj4[q0];     *(float4*)&hjV[4]  = hj4[q0 + 1];
        *(float4*)&hjV[8]  = hj4[q0 + 4]; *(float4*)&hjV[12] = hj4[q0 + 5];
        *(float4*)&wdV[0]  = wd4[q0];     *(float4*)&wdV[4]  = wd4[q0 + 1];
        *(float4*)&wdV[8]  = wd4[q0 + 4]; *(float4*)&wdV[12] = wd4[q0 + 5];
        *(float4*)&b1V[0]  = bb4[q0];     *(float4*)&b1V[4]  = bb4[q0 + 1];
        *(float4*)&b1V[8]  = bb4[q0 + 4]; *(float4*)&b1V[12] = bb4[q0 + 5];

        Frag16 fa, fb;
#pragma unroll
        for (int e = 0; e < 16; ++e) {
            float h = silu_f(hiV[e] + hjV[e] + dist * wdV[e] + b1V[e]);
            fa.v[e] = (__bf16)h;
        }
        *(uint4*)&fb.u[0] = *(const uint4*)&sBt[n16 * EHP + k0 + kbu];
        *(uint4*)&fb.u[4] = *(const uint4*)&sBt[n16 * EHP + k0 + kbu + 16];
        acc = __builtin_amdgcn_wmma_f32_16x16x32_bf16(false, fa.v, false, fb.v,
                                                      (short)0, acc, false, false);
    }

    float bn = b_e2[n16];
#pragma unroll
    for (int r = 0; r < 8; ++r) {
        int eloc  = r + ((lane >> 4) << 3);
        int edge2 = e0 + eloc;
        int node2 = edge2 >> 5;
        int b2    = node2 >> 12;
        int j2    = g_idx[edge2];
        int pm    = mask[node2] & mask[b2 * NN + j2];
        float v = silu_f(acc[r] + bn);
        m_out[(size_t)edge2 * 16 + n16] = pm ? v : 0.0f;
    }
}

// ============================================================================
// Kernel 4: coor MLP (tiny: 16->64->1) per edge; pair-masked weight.
// ============================================================================
__global__ void egnn_coorw(const float* __restrict__ m_ij,
                           const int* __restrict__ g_idx,
                           const unsigned char* __restrict__ mask,
                           const float* __restrict__ w_c1, const float* __restrict__ b_c1,
                           const float* __restrict__ w_c2, const float* __restrict__ b_c2,
                           float* __restrict__ coorw)
{
    __shared__ float sW[16 * 64];
    __shared__ float sb[64];
    __shared__ float sw2[64];
    int tid = threadIdx.x;
    for (int e = tid; e < 16 * 64; e += 256) sW[e] = w_c1[e];
    if (tid < 64) { sb[tid] = b_c1[tid]; sw2[tid] = w_c2[tid]; }
    __syncthreads();

    int edge = blockIdx.x * 256 + tid;
    float m[16];
#pragma unroll
    for (int t2 = 0; t2 < 16; ++t2) m[t2] = m_ij[(size_t)edge * 16 + t2];
    float accv = b_c2[0];
    for (int o = 0; o < 64; ++o) {
        float s = sb[o];
#pragma unroll
        for (int t2 = 0; t2 < 16; ++t2) s += m[t2] * sW[t2 * 64 + o];
        accv += silu_f(s) * sw2[o];
    }
    int node = edge >> 5;
    int b    = node >> 12;
    int j    = g_idx[edge];
    int pm   = mask[node] & mask[b * NN + j];
    coorw[edge] = pm ? accv : 0.0f;
}

// ============================================================================
// Kernel 5: coordinate update
// ============================================================================
__global__ void egnn_coor_out(const float* __restrict__ coors,
                              const int* __restrict__ g_idx,
                              const float* __restrict__ coorw,
                              float* __restrict__ out_coors)
{
    int node = blockIdx.x * 256 + threadIdx.x;
    if (node >= BB * NN) return;
    int b = node >> 12;
    int i = node & (NN - 1);
    const float* cb = coors + (size_t)b * NN * 3;
    float xi = cb[i * 3 + 0], yi = cb[i * 3 + 1], zi = cb[i * 3 + 2];
    float ax = xi, ay = yi, az = zi;
    for (int k = 0; k < KKN; ++k) {
        int edge = node * KKN + k;
        int j = g_idx[edge];
        float w = coorw[edge];
        ax += w * (xi - cb[j * 3 + 0]);
        ay += w * (yi - cb[j * 3 + 1]);
        az += w * (zi - cb[j * 3 + 2]);
    }
    out_coors[node * 3 + 0] = ax;
    out_coors[node * 3 + 1] = ay;
    out_coors[node * 3 + 2] = az;
}

// ============================================================================
// Kernel 6: LayerNorm(feats) + m_i pooling -> node_in [B*N x 160]
// ============================================================================
__global__ void egnn_ln_msum(const float* __restrict__ feats,
                             const float* __restrict__ m_ij,
                             const float* __restrict__ ln_g, const float* __restrict__ ln_b,
                             float* __restrict__ node_in)
{
    __shared__ float red[128];
    int node = blockIdx.x;
    int t = threadIdx.x;
    float x = feats[(size_t)node * DD + t];
    red[t] = x; __syncthreads();
    for (int s = 64; s > 0; s >>= 1) { if (t < s) red[t] += red[t + s]; __syncthreads(); }
    float mu = red[0] / (float)DD;
    __syncthreads();
    float dx = x - mu;
    red[t] = dx * dx; __syncthreads();
    for (int s = 64; s > 0; s >>= 1) { if (t < s) red[t] += red[t + s]; __syncthreads(); }
    float var = red[0] / (float)DD;
    float y = dx * rsqrtf(var + 1e-5f) * ln_g[t] + ln_b[t];
    node_in[(size_t)node * NIP + t] = y;
    if (t < MMO) {
        float s = 0.0f;
        for (int k = 0; k < KKN; ++k) s += m_ij[((size_t)node * KKN + k) * 16 + t];
        node_in[(size_t)node * NIP + DD + t] = s;
        node_in[(size_t)node * NIP + DD + MMO + t] = 0.0f;
    }
}

// ============================================================================
// launch
// ============================================================================
extern "C" void kernel_launch(void* const* d_in, const int* in_sizes, int n_in,
                              void* d_out, int out_size, void* d_ws, size_t ws_size,
                              hipStream_t stream)
{
    (void)in_sizes; (void)n_in; (void)out_size; (void)ws_size;

    const float* feats = (const float*)d_in[0];
    const float* coors = (const float*)d_in[1];
    const unsigned char* mask = (const unsigned char*)d_in[2];
    const float* w_e1 = (const float*)d_in[3];
    const float* b_e1 = (const float*)d_in[4];
    const float* w_e2 = (const float*)d_in[5];
    const float* b_e2 = (const float*)d_in[6];
    const float* w_c1 = (const float*)d_in[7];
    const float* b_c1 = (const float*)d_in[8];
    const float* w_c2 = (const float*)d_in[9];
    const float* b_c2 = (const float*)d_in[10];
    const float* w_n1 = (const float*)d_in[11];
    const float* b_n1 = (const float*)d_in[12];
    const float* w_n2 = (const float*)d_in[13];
    const float* b_n2 = (const float*)d_in[14];
    const float* ln_g = (const float*)d_in[15];
    const float* ln_b = (const float*)d_in[16];

    char* ws = (char*)d_ws;
    int*    g_idx   = (int*)   (ws + ((size_t)0  << 20));  // 1 MiB
    float*  g_dist  = (float*) (ws + ((size_t)1  << 20));  // 1 MiB
    float*  coorw   = (float*) (ws + ((size_t)2  << 20));  // 1 MiB
    float*  m_ij    = (float*) (ws + ((size_t)3  << 20));  // 16 MiB
    float*  Hi      = (float*) (ws + ((size_t)19 << 20));  // 18 MiB
    float*  Hj      = (float*) (ws + ((size_t)37 << 20));  // 18 MiB
    float*  node_in = (float*) (ws + ((size_t)55 << 20));  // 5 MiB
    float*  hid     = (float*) (ws + ((size_t)60 << 20));  // 8 MiB
    __bf16* bt_top  = (__bf16*)(ws + ((size_t)68 << 20));
    __bf16* bt_mid  = (__bf16*)(ws + ((size_t)68 << 20) + ((size_t)256 << 10));
    __bf16* bt_n1   = (__bf16*)(ws + ((size_t)68 << 20) + ((size_t)512 << 10));
    __bf16* bt_n2   = (__bf16*)(ws + ((size_t)68 << 20) + ((size_t)768 << 10));

    float* out_nodes = (float*)d_out;                       // [B*N*D]
    float* out_coors = out_nodes + (size_t)BB * NN * DD;    // [B*N*3]

    // 0) weight prep: transpose + pad + bf16 (once; all tiny)
    egnn_prep_bt<<<(DD * HSTR + 255) / 256, 256, 0, stream>>>(
        w_e1, DD, EH, EH, bt_top, DD, HSTR);
    egnn_prep_bt<<<(DD * HSTR + 255) / 256, 256, 0, stream>>>(
        w_e1 + (size_t)DD * EH, DD, EH, EH, bt_mid, DD, HSTR);
    egnn_prep_bt<<<(NIP * 2 * DD + 255) / 256, 256, 0, stream>>>(
        w_n1, 144, 2 * DD, 2 * DD, bt_n1, NIP, 2 * DD);
    egnn_prep_bt<<<(2 * DD * DD + 255) / 256, 256, 0, stream>>>(
        w_n2, 2 * DD, DD, DD, bt_n2, 2 * DD, DD);

    // 1) top-K neighbors
    egnn_topk<<<BB * NN, 256, 0, stream>>>(coors, mask, g_idx, g_dist);

    // 2) Hi / Hj projections (WMMA), rows stride 576, pad cols zeroed
    egnn_gemm_bf16<<<dim3(HSTR / 64, (BB * NN) / 16), 128, 0, stream>>>(
        feats, DD, bt_top, DD, Hi, HSTR, BB * NN, EH, DD, nullptr, nullptr, 0);
    egnn_gemm_bf16<<<dim3(HSTR / 64, (BB * NN) / 16), 128, 0, stream>>>(
        feats, DD, bt_mid, DD, Hj, HSTR, BB * NN, EH, DD, nullptr, nullptr, 0);

    // 3) fused edge MLP -> m_ij (WMMA)
    egnn_edge<<<(BB * NN * KKN / 16) / 8, 256, 0, stream>>>(
        Hi, Hj, g_idx, g_dist, mask, w_e1, b_e1, w_e2, b_e2, m_ij);

    // 4) coor weights + coordinate output
    egnn_coorw<<<(BB * NN * KKN) / 256, 256, 0, stream>>>(
        m_ij, g_idx, mask, w_c1, b_c1, w_c2, b_c2, coorw);
    egnn_coor_out<<<(BB * NN + 255) / 256, 256, 0, stream>>>(coors, g_idx, coorw, out_coors);

    // 5) LN + message pooling -> node_in [B*N x 160]
    egnn_ln_msum<<<BB * NN, 128, 0, stream>>>(feats, m_ij, ln_g, ln_b, node_in);

    // 6) node MLP (WMMA)
    egnn_gemm_bf16<<<dim3((2 * DD) / 64, (BB * NN) / 16), 128, 0, stream>>>(
        node_in, NIP, bt_n1, NIP, hid, 2 * DD, BB * NN, 2 * DD, NIP, b_n1, nullptr, 1);
    egnn_gemm_bf16<<<dim3(DD / 64, (BB * NN) / 16), 128, 0, stream>>>(
        hid, 2 * DD, bt_n2, 2 * DD, out_nodes, DD, BB * NN, DD, 2 * DD, b_n2, feats, 0);
}
